// RUNetContext_56667798503491
// MI455X (gfx1250) — compile-verified, matrix operand order checked by reading, not run on hardware
//
#include <hip/hip_runtime.h>
#include <hip/hip_bf16.h>

// ---------------- CDNA5 WMMA types ----------------
typedef __bf16 v16bf __attribute__((ext_vector_type(16)));
typedef float  v8f   __attribute__((ext_vector_type(8)));
typedef unsigned int v8u __attribute__((ext_vector_type(8)));

#define B_  8
#define N_  128
#define P_  2048
#define H_  512
#define NCLS_ 151
#define DOBJ_ 4096
#define DCAT_ 4424   // 4096 + 200 + 128

__device__ __forceinline__ unsigned int pk_bf16(float a, float b) {
  unsigned int ua = __float_as_uint(a);
  unsigned int ub = __float_as_uint(b);
  ua = (ua + 0x7FFFu + ((ua >> 16) & 1u)) >> 16;
  ub = (ub + 0x7FFFu + ((ub >> 16) & 1u)) >> 16;
  return (ua & 0xFFFFu) | (ub << 16);
}
__device__ __forceinline__ unsigned short f2bf(float a) {
  unsigned int ua = __float_as_uint(a);
  return (unsigned short)((ua + 0x7FFFu + ((ua >> 16) & 1u)) >> 16);
}

// ---------------- staging helpers (128x32 tiles, 256 threads) ----------------

__device__ __forceinline__ void stageA_fast(const float* __restrict__ A,
                                            unsigned int (*__restrict__ As)[17],
                                            int tid, int m0, int k0, int K)
{
  const int row  = tid >> 1;
  const int colq = (tid & 1) << 4;
  const float* ap = A + (long long)(m0 + row) * K + k0 + colq;
#pragma unroll
  for (int h = 0; h < 4; ++h) {
    const float4 v = *(const float4*)(ap + h * 4);
    const int cp = (colq >> 1) + h * 2;
    As[row][cp + 0] = pk_bf16(v.x, v.y);
    As[row][cp + 1] = pk_bf16(v.z, v.w);
  }
}

// branchless clamped loads + cndmask-zero (no exec-mask branch chains)
__device__ __forceinline__ void stageA_guard(const float* __restrict__ A,
                                             unsigned int (*__restrict__ As)[17],
                                             int tid, int m0, int k0, int M, int K)
{
  const int row  = tid >> 1;
  const int colq = (tid & 1) << 4;
  const int gr = m0 + row;
  const int cr = gr < M ? gr : M - 1;
  const float* ap = A + (long long)cr * K;
  float tmp[16];
#pragma unroll
  for (int j = 0; j < 16; ++j) {
    const int gk = k0 + colq + j;
    const int ck = gk < K ? gk : K - 1;
    const float v = ap[ck];
    tmp[j] = (gr < M && gk < K) ? v : 0.f;
  }
#pragma unroll
  for (int h = 0; h < 8; ++h)
    As[row][(colq >> 1) + h] = pk_bf16(tmp[2 * h], tmp[2 * h + 1]);
}

// B given K x N row-major -> Bs[n][k] via coalesced float4 along N
__device__ __forceinline__ void stageBkN_fast(const float* __restrict__ Bm,
                                              unsigned int (*__restrict__ Bs)[17],
                                              int tid, int n0, int k0, int N)
{
  const int krow = tid >> 3;
  const int nq   = (tid & 7) << 4;
  unsigned short* bsp = (unsigned short*)Bs;
  const float* bp = Bm + (long long)(k0 + krow) * N + n0 + nq;
#pragma unroll
  for (int h = 0; h < 4; ++h) {
    const float4 v = *(const float4*)(bp + h * 4);
    const int nb = nq + h * 4;
    bsp[(nb + 0) * 34 + krow] = f2bf(v.x);
    bsp[(nb + 1) * 34 + krow] = f2bf(v.y);
    bsp[(nb + 2) * 34 + krow] = f2bf(v.z);
    bsp[(nb + 3) * 34 + krow] = f2bf(v.w);
  }
}

__device__ __forceinline__ void stageBkN_guard(const float* __restrict__ Bm,
                                               unsigned int (*__restrict__ Bs)[17],
                                               int tid, int n0, int k0, int N, int K)
{
  const int krow = tid >> 3;
  const int nq   = (tid & 7) << 4;
  const int gk = k0 + krow;
  const int ck = gk < K ? gk : K - 1;
  const bool kin = gk < K;
  const float* bp = Bm + (long long)ck * N;
  unsigned short* bsp = (unsigned short*)Bs;
#pragma unroll
  for (int j = 0; j < 16; ++j) {
    const int gn = n0 + nq + j;
    const int cn = gn < N ? gn : N - 1;
    const float v = bp[cn];
    bsp[(nq + j) * 34 + krow] = f2bf((kin && gn < N) ? v : 0.f);
  }
}

// B given N x K row-major (transB): straight K-run copy
__device__ __forceinline__ void stageBtK_fast(const float* __restrict__ Bm,
                                              unsigned int (*__restrict__ Bs)[17],
                                              int tid, int n0, int k0, int K)
{
  const int nrow = tid >> 1;
  const int colq = (tid & 1) << 4;
  const float* bp = Bm + (long long)(n0 + nrow) * K + k0 + colq;
#pragma unroll
  for (int h = 0; h < 4; ++h) {
    const float4 v = *(const float4*)(bp + h * 4);
    const int cp = (colq >> 1) + h * 2;
    Bs[nrow][cp + 0] = pk_bf16(v.x, v.y);
    Bs[nrow][cp + 1] = pk_bf16(v.z, v.w);
  }
}

__device__ __forceinline__ void stageBtK_guard(const float* __restrict__ Bm,
                                               unsigned int (*__restrict__ Bs)[17],
                                               int tid, int n0, int k0, int N, int K)
{
  const int nrow = tid >> 1;
  const int colq = (tid & 1) << 4;
  const int gn = n0 + nrow;
  const int cn = gn < N ? gn : N - 1;
  const float* bp = Bm + (long long)cn * K;
  float tmp[16];
#pragma unroll
  for (int j = 0; j < 16; ++j) {
    const int gk = k0 + colq + j;
    const int ck = gk < K ? gk : K - 1;
    const float v = bp[ck];
    tmp[j] = (gn < N && gk < K) ? v : 0.f;
  }
#pragma unroll
  for (int h = 0; h < 8; ++h)
    Bs[nrow][(colq >> 1) + h] = pk_bf16(tmp[2 * h], tmp[2 * h + 1]);
}

// fragment gather per ISA 16-bit A layout + 2x4 WMMA
__device__ __forceinline__ void frag_mma(const unsigned int (*__restrict__ As)[17],
                                         const unsigned int (*__restrict__ Bs)[17],
                                         int wm, int wn, int hi, int lr,
                                         v8f acc[2][4])
{
  v8u af[2], bf[4];
#pragma unroll
  for (int s = 0; s < 2; ++s)
#pragma unroll
    for (int v = 0; v < 8; ++v) {
      const int pair = (v < 4 ? v : v + 4) + hi * 4;
      af[s][v] = As[wm * 32 + s * 16 + lr][pair];
    }
#pragma unroll
  for (int t = 0; t < 4; ++t)
#pragma unroll
    for (int v = 0; v < 8; ++v) {
      const int pair = (v < 4 ? v : v + 4) + hi * 4;
      bf[t][v] = Bs[wn * 64 + t * 16 + lr][pair];
    }
#pragma unroll
  for (int s = 0; s < 2; ++s)
#pragma unroll
    for (int t = 0; t < 4; ++t)
      acc[s][t] = __builtin_amdgcn_wmma_f32_16x16x32_bf16(
          false, __builtin_bit_cast(v16bf, af[s]),
          false, __builtin_bit_cast(v16bf, bf[t]),
          (short)0, acc[s][t], false, false);
}

// =====================================================================
// Generic strided-batch GEMM: C = act( A(MxK) @ B(KxN or NxK^T) + bias
//                                      [+ addsrc] )   via bf16 WMMA.
// Block = 256 threads = 8 waves.  Tile 128(M) x 128(N), K step 32.
// Steady-state loop has NO bounds logic: guards hoisted to a separate
// (cold) loop for edge tiles and a single peeled K-tail step.
// =====================================================================
__global__ __launch_bounds__(256, 1)
void k_gemm(const float* __restrict__ A, const float* __restrict__ Bm,
            const float* __restrict__ bias, const float* __restrict__ addsrc,
            float* __restrict__ C,
            int M, int N, int K,
            long long sA, long long sB, long long sC, long long sAdd,
            int transB, int relu)
{
  __shared__ unsigned int As[128][17];  // [m][kpair]
  __shared__ unsigned int Bs[128][17];  // [n][kpair] (B staged transposed)

  const int bz = blockIdx.z;
  A  += (long long)bz * sA;
  Bm += (long long)bz * sB;
  C  += (long long)bz * sC;
  const float* add = addsrc ? addsrc + (long long)bz * sAdd : nullptr;

  const int m0 = blockIdx.y * 128;
  const int n0 = blockIdx.x * 128;

  const int tid  = threadIdx.x;
  const int lane = tid & 31;
  const int wave = tid >> 5;
  const int wm = wave >> 1;   // 0..3  (32-row strip)
  const int wn = wave & 1;    // 0..1  (64-col strip)
  const int hi = lane >> 4;
  const int lr = lane & 15;

  const bool interiorM = (m0 + 128 <= M);
  const bool interiorN = (n0 + 128 <= N);
  const bool fast = interiorM && interiorN && (transB || ((N & 3) == 0));

  v8f acc[2][4];
#pragma unroll
  for (int s = 0; s < 2; ++s)
#pragma unroll
    for (int t = 0; t < 4; ++t) acc[s][t] = (v8f){};

  const int nkf = K >> 5;          // full 32-wide K tiles
  const int ktail = K & 31;

  if (fast) {
    // -------- steady state: zero guards --------
    for (int kt = 0; kt < nkf; ++kt) {
      const int k0 = kt << 5;
      stageA_fast(A, As, tid, m0, k0, K);
      if (transB) stageBtK_fast(Bm, Bs, tid, n0, k0, K);
      else        stageBkN_fast(Bm, Bs, tid, n0, k0, N);
      if (kt + 1 < nkf) {   // speculative next-tile prefetch (OOB-safe)
        __builtin_prefetch(A + (long long)(m0 + (tid >> 1)) * K + k0 + 32, 0, 1);
        if (transB)
          __builtin_prefetch(Bm + (long long)(n0 + (tid >> 1)) * K + k0 + 32, 0, 1);
        else
          __builtin_prefetch(Bm + (long long)(k0 + 32 + (tid >> 3)) * N + n0, 0, 1);
      }
      __syncthreads();
      frag_mma(As, Bs, wm, wn, hi, lr, acc);
      __syncthreads();
    }
  } else {
    // -------- edge tiles: branchless clamped staging --------
    for (int kt = 0; kt < nkf; ++kt) {
      const int k0 = kt << 5;
      stageA_guard(A, As, tid, m0, k0, M, K);
      if (transB) stageBtK_guard(Bm, Bs, tid, n0, k0, N, K);
      else        stageBkN_guard(Bm, Bs, tid, n0, k0, N, K);
      __syncthreads();
      frag_mma(As, Bs, wm, wn, hi, lr, acc);
      __syncthreads();
    }
  }

  if (ktail) {
    const int k0 = nkf << 5;
    stageA_guard(A, As, tid, m0, k0, M, K);
    if (transB) stageBtK_guard(Bm, Bs, tid, n0, k0, N, K);
    else        stageBkN_guard(Bm, Bs, tid, n0, k0, N, K);
    __syncthreads();
    frag_mma(As, Bs, wm, wn, hi, lr, acc);
    __syncthreads();
  }

  // ---------- epilogue: D layout (lanes0-15 M=v, lanes16-31 M=v+8) ----------
  if (interiorM && interiorN) {
#pragma unroll
    for (int s = 0; s < 2; ++s)
#pragma unroll
      for (int t = 0; t < 4; ++t) {
        const int gc = n0 + wn * 64 + t * 16 + lr;
#pragma unroll
        for (int v = 0; v < 8; ++v) {
          const int gr = m0 + wm * 32 + s * 16 + v + hi * 8;
          float x = acc[s][t][v];
          if (bias) x += bias[gc];
          if (add)  x += add[(long long)gr * N + gc];
          if (relu) x = fmaxf(x, 0.f);
          C[(long long)gr * N + gc] = x;
        }
      }
  } else {
#pragma unroll
    for (int s = 0; s < 2; ++s)
#pragma unroll
      for (int t = 0; t < 4; ++t) {
        const int gc = n0 + wn * 64 + t * 16 + lr;
        if (gc >= N) continue;
#pragma unroll
        for (int v = 0; v < 8; ++v) {
          const int gr = m0 + wm * 32 + s * 16 + v + hi * 8;
          if (gr >= M) continue;
          float x = acc[s][t][v];
          if (bias) x += bias[gc];
          if (add)  x += add[(long long)gr * N + gc];
          if (relu) x = fmaxf(x, 0.f);
          C[(long long)gr * N + gc] = x;
        }
      }
  }
}

// ======================= small kernels =======================

// softmax(logits)@obj_embed, pos MLP, concat with roi -> Xcat (1024 x 4424)
__global__ __launch_bounds__(256)
void k_embed_concat(const float* __restrict__ roi, const float* __restrict__ logits,
                    const float* __restrict__ bboxes, const float* __restrict__ obj_embed,
                    const float* __restrict__ pos_w, const float* __restrict__ pos_b,
                    float* __restrict__ Xcat)
{
  const int row = blockIdx.x;       // 0..1023
  const int tid = threadIdx.x;
  __shared__ float red[256];
  __shared__ float p[NCLS_];

  const float* lg = logits + (long long)row * NCLS_;
  float v = (tid < NCLS_) ? lg[tid] : -1e30f;
  red[tid] = v; __syncthreads();
  for (int st = 128; st > 0; st >>= 1) {
    if (tid < st) red[tid] = fmaxf(red[tid], red[tid + st]);
    __syncthreads();
  }
  const float mx = red[0]; __syncthreads();
  float e = (tid < NCLS_) ? __expf(v - mx) : 0.f;
  red[tid] = e; __syncthreads();
  for (int st = 128; st > 0; st >>= 1) {
    if (tid < st) red[tid] += red[tid + st];
    __syncthreads();
  }
  const float s = red[0]; __syncthreads();
  if (tid < NCLS_) p[tid] = e / s;
  __syncthreads();

  float* xr = Xcat + (long long)row * DCAT_;
  for (int j = tid; j < DOBJ_; j += 256)
    xr[j] = roi[(long long)row * DOBJ_ + j];
  for (int j = tid; j < 200; j += 256) {
    float acc = 0.f;
    for (int c = 0; c < NCLS_; ++c) acc += p[c] * obj_embed[c * 200 + j];
    xr[DOBJ_ + j] = acc;
  }
  const float inv = rsqrtf(1.0f + 1e-5f);
  const float* bb = bboxes + (long long)row * 5;
  for (int j = tid; j < 128; j += 256) {
    float acc = pos_b[j];
    for (int c = 0; c < 5; ++c) acc += bb[c] * inv * pos_w[c * 128 + j];
    xr[DOBJ_ + 200 + j] = fmaxf(acc, 0.f);
  }
}

// per-row layernorm (H=512)
__global__ __launch_bounds__(128)
void k_ln(const float* __restrict__ x, const float* __restrict__ g,
          const float* __restrict__ bta, float* __restrict__ y)
{
  const int row = blockIdx.x;
  const int tid = threadIdx.x;
  __shared__ float r1[128], r2[128];
  const float* xr = x + (long long)row * H_;
  float s = 0.f, ss = 0.f;
  for (int j = tid; j < H_; j += 128) { float v = xr[j]; s += v; ss += v * v; }
  r1[tid] = s; r2[tid] = ss; __syncthreads();
  for (int st = 64; st > 0; st >>= 1) {
    if (tid < st) { r1[tid] += r1[tid + st]; r2[tid] += r2[tid + st]; }
    __syncthreads();
  }
  const float mean = r1[0] * (1.0f / H_);
  const float var = r2[0] * (1.0f / H_) - mean * mean;
  const float inv = rsqrtf(var + 1e-5f);
  float* yr = y + (long long)row * H_;
  for (int j = tid; j < H_; j += 128)
    yr[j] = (xr[j] - mean) * inv * g[j] + bta[j];
}

// row squared norms of obj (1024 rows)
__global__ __launch_bounds__(128)
void k_rownorm(const float* __restrict__ x, float* __restrict__ r)
{
  const int row = blockIdx.x;
  const int tid = threadIdx.x;
  __shared__ float red[128];
  const float* xr = x + (long long)row * H_;
  float s = 0.f;
  for (int j = tid; j < H_; j += 128) { float v = xr[j]; s += v * v; }
  red[tid] = s; __syncthreads();
  for (int st = 64; st > 0; st >>= 1) {
    if (tid < st) red[tid] += red[tid + st];
    __syncthreads();
  }
  if (tid == 0) r[row] = red[0];
}

__global__ void k_at_init(float* __restrict__ at)
{
  const int idx = blockIdx.x * blockDim.x + threadIdx.x;
  if (idx >= B_ * N_ * N_) return;
  const int rem = idx & (N_ * N_ - 1);
  const int i = rem >> 7, j = rem & (N_ - 1);
  at[idx] = (i == j) ? -10000.0f : 0.0f;
}

// af = dot(s[src]*o[dst]*phr, w_w)+w_b -> scatter-add into at[b,src,dst]
__global__ __launch_bounds__(128)
void k_edge(const float* __restrict__ s, const float* __restrict__ o,
            const float* __restrict__ phr, const int* __restrict__ pairs,
            const float* __restrict__ ww, const float* __restrict__ wb,
            float* __restrict__ at)
{
  const int bp = blockIdx.x;            // 0..B*P-1
  const int b = bp >> 11;               // P=2048
  const int src = pairs[bp * 2 + 0];
  const int dst = pairs[bp * 2 + 1];
  const float* sv = s + ((long long)(b * N_ + src)) * H_;
  const float* ov = o + ((long long)(b * N_ + dst)) * H_;
  const float* uv = phr + (long long)bp * H_;
  float acc = 0.f;
  for (int h = threadIdx.x; h < H_; h += 128)
    acc += sv[h] * ov[h] * uv[h] * ww[h];
  __shared__ float red[128];
  red[threadIdx.x] = acc; __syncthreads();
  for (int st = 64; st > 0; st >>= 1) {
    if (threadIdx.x < st) red[threadIdx.x] += red[threadIdx.x + st];
    __syncthreads();
  }
  if (threadIdx.x == 0)
    atomicAdd(&at[((long long)b * N_ + src) * N_ + dst], red[0] + wb[0]);
}

// amap[b,i,:] = omega(norm) * softmax(at[b,i,:])
__global__ __launch_bounds__(128)
void k_amap(const float* __restrict__ at, const float* __restrict__ G,
            const float* __restrict__ r, float* __restrict__ amap)
{
  const int bi = blockIdx.x;            // 0..1023
  const int b = bi >> 7, i = bi & (N_ - 1);
  const int j = threadIdx.x;
  __shared__ float red[128];
  const long long base = ((long long)b * N_ + i) * N_;
  const float a = at[base + j];
  red[j] = a; __syncthreads();
  for (int st = 64; st > 0; st >>= 1) {
    if (j < st) red[j] = fmaxf(red[j], red[j + st]);
    __syncthreads();
  }
  const float mx = red[0]; __syncthreads();
  const float e = __expf(a - mx);
  red[j] = e; __syncthreads();
  for (int st = 64; st > 0; st >>= 1) {
    if (j < st) red[j] += red[j + st];
    __syncthreads();
  }
  const float sm = e / red[0];
  // dist from Gram matrix
  const float d2 = fmaxf(r[b * N_ + i] + r[b * N_ + j] - 2.0f * G[base + j], 0.f);
  const float nrm = sqrtf(d2);
  float omega;
  if (i == j) omega = 0.f;
  else if (nrm < 0.5f) omega = 4.0f;
  else if (nrm < 1.0f) { const float c = fmaxf(nrm, 1e-5f); omega = 1.0f / (c * c); }
  else omega = 0.f;
  amap[base + j] = omega * sm;
}

// preds = argmax(dists[:,1:]) + 1 (softmax is monotonic)
__global__ void k_pred(const float* __restrict__ dists, float* __restrict__ preds)
{
  const int row = blockIdx.x * blockDim.x + threadIdx.x;
  if (row >= B_ * N_) return;
  const float* d = dists + (long long)row * NCLS_;
  int best = 1; float bv = d[1];
  for (int j = 2; j < NCLS_; ++j)
    if (d[j] > bv) { bv = d[j]; best = j; }
  preds[row] = (float)best;
}

// ======================= host side =======================

static void gemm(hipStream_t st, const float* A, const float* Bm,
                 const float* bias, const float* addsrc, float* C,
                 int M, int N, int K, int batch,
                 long long sA, long long sB, long long sC, long long sAdd,
                 int transB, int relu)
{
  dim3 g((N + 127) / 128, (M + 127) / 128, batch);
  k_gemm<<<g, 256, 0, st>>>(A, Bm, bias, addsrc, C, M, N, K,
                            sA, sB, sC, sAdd, transB, relu);
}

extern "C" void kernel_launch(void* const* d_in, const int* in_sizes, int n_in,
                              void* d_out, int out_size, void* d_ws, size_t ws_size,
                              hipStream_t stream)
{
  const float* roi    = (const float*)d_in[0];
  const float* logits = (const float*)d_in[1];
  const float* bboxes = (const float*)d_in[2];
  const float* unionf = (const float*)d_in[3];
  const int*   pairs  = (const int*)d_in[4];
  const float* obj_embed = (const float*)d_in[5];
  const float* pos_w = (const float*)d_in[6];
  const float* pos_b = (const float*)d_in[7];
  const float* merge_w = (const float*)d_in[8];
  const float* merge_b = (const float*)d_in[9];
  const float* phr_w = (const float*)d_in[10];
  const float* phr_b = (const float*)d_in[11];
  const float* cls_w = (const float*)d_in[12];
  const float* cls_b = (const float*)d_in[13];
  const float* ln_g = (const float*)d_in[14];
  const float* ln_b = (const float*)d_in[15];

  const int ROWS = B_ * N_;          // 1024
  const int PR   = B_ * P_;          // 16384

  float* ws = (float*)d_ws;
  float* Xcat = ws;                  ws += (long long)ROWS * DCAT_;   // 4,530,176
  float* objA = ws;                  ws += (long long)ROWS * H_;
  float* objB = ws;                  ws += (long long)ROWS * H_;
  float* phr  = ws;                  ws += (long long)PR * H_;        // 8,388,608
  float* sbuf = ws;                  ws += (long long)ROWS * H_;
  float* obuf = ws;                  ws += (long long)ROWS * H_;
  float* atb  = ws;                  ws += (long long)B_ * N_ * N_;
  float* Gb   = ws;                  ws += (long long)B_ * N_ * N_;
  float* amap = ws;                  ws += (long long)B_ * N_ * N_;
  float* rbuf = ws;                  ws += ROWS;
  float* tmpA = ws;                  ws += (long long)ROWS * H_;
  float* tmpB = ws;                  ws += (long long)ROWS * H_;
  float* tmpC = ws;                  ws += (long long)ROWS * 2 * H_;

  // 1) embed/pos/concat
  k_embed_concat<<<ROWS, 256, 0, stream>>>(roi, logits, bboxes, obj_embed,
                                           pos_w, pos_b, Xcat);
  // 2) obj_feats = Xcat @ merge_w + merge_b
  gemm(stream, Xcat, merge_w, merge_b, nullptr, objA,
       ROWS, H_, DCAT_, 1, 0, 0, 0, 0, 0, 0);
  // 3) phr = union @ phr_w + phr_b  (dominant GEMM, 68.7 GFLOP)
  gemm(stream, unionf, phr_w, phr_b, nullptr, phr,
       PR, H_, DOBJ_, 1, 0, 0, 0, 0, 0, 0);

  // 4) two message-passing blocks
  for (int blk = 0; blk < 2; ++blk) {
    const int pb = 16 + blk * 16;
    const float* ws_w   = (const float*)d_in[pb + 0];
    const float* ws_b   = (const float*)d_in[pb + 1];
    const float* wo_w   = (const float*)d_in[pb + 2];
    const float* wo_b   = (const float*)d_in[pb + 3];
    const float* w_w    = (const float*)d_in[pb + 4];
    const float* w_b    = (const float*)d_in[pb + 5];
    const float* conv_w = (const float*)d_in[pb + 6];
    const float* conv_b = (const float*)d_in[pb + 7];
    const float* t1_w   = (const float*)d_in[pb + 8];
    const float* t1_b   = (const float*)d_in[pb + 9];
    const float* t2_w   = (const float*)d_in[pb + 10];
    const float* t2_b   = (const float*)d_in[pb + 11];
    const float* ln1_g  = (const float*)d_in[pb + 12];
    const float* ln1_b  = (const float*)d_in[pb + 13];
    const float* ln2_g  = (const float*)d_in[pb + 14];
    const float* ln2_b  = (const float*)d_in[pb + 15];

    // pairwise distances via Gram matrix: G = obj @ obj^T (batched, transB)
    k_rownorm<<<ROWS, 128, 0, stream>>>(objA, rbuf);
    gemm(stream, objA, objA, nullptr, nullptr, Gb,
         N_, N_, H_, B_, (long long)N_ * H_, (long long)N_ * H_,
         (long long)N_ * N_, 0, 1, 0);
    // s, o projections
    gemm(stream, objA, ws_w, ws_b, nullptr, sbuf, ROWS, H_, H_, 1, 0,0,0,0, 0, 0);
    gemm(stream, objA, wo_w, wo_b, nullptr, obuf, ROWS, H_, H_, 1, 0,0,0,0, 0, 0);
    // attention logits with scatter-add
    k_at_init<<<(B_ * N_ * N_ + 255) / 256, 256, 0, stream>>>(atb);
    k_edge<<<PR, 128, 0, stream>>>(sbuf, obuf, phr, pairs, w_w, w_b, atb);
    k_amap<<<ROWS, 128, 0, stream>>>(atb, Gb, rbuf, amap);
    // conv = relu(LN1(obj) @ conv_w + conv_b)
    k_ln<<<ROWS, 128, 0, stream>>>(objA, ln1_g, ln1_b, tmpA);
    gemm(stream, tmpA, conv_w, conv_b, nullptr, tmpB, ROWS, H_, H_, 1, 0,0,0,0, 0, 1);
    // out = obj + amap @ conv  (batched)
    gemm(stream, amap, tmpB, nullptr, objA, objB,
         N_, H_, N_, B_, (long long)N_ * N_, (long long)N_ * H_,
         (long long)N_ * H_, (long long)N_ * H_, 0, 0);
    // MLP: obj = relu(out + relu(LN2(out)@t1+b1)@t2+b2)
    k_ln<<<ROWS, 128, 0, stream>>>(objB, ln2_g, ln2_b, tmpA);
    gemm(stream, tmpA, t1_w, t1_b, nullptr, tmpC, ROWS, 2 * H_, H_, 1, 0,0,0,0, 0, 1);
    gemm(stream, tmpC, t2_w, t2_b, objB, objA, ROWS, H_, 2 * H_, 1, 0,0,0,0, 0, 1);
  }

  // 5) classifier + outputs
  float* dout  = (float*)d_out;
  float* preds = dout + (long long)ROWS * NCLS_;
  float* feats = preds + ROWS;
  k_ln<<<ROWS, 128, 0, stream>>>(objA, ln_g, ln_b, tmpA);
  gemm(stream, tmpA, cls_w, cls_b, nullptr, dout, ROWS, NCLS_, H_, 1, 0,0,0,0, 0, 0);
  k_pred<<<(ROWS + 255) / 256, 256, 0, stream>>>(dout, preds);
  hipMemcpyAsync(feats, objA, (size_t)ROWS * H_ * sizeof(float),
                 hipMemcpyDeviceToDevice, stream);

  (void)in_sizes; (void)n_in; (void)out_size; (void)ws_size;
}
